// MultiHeadAttention_79748952752215
// MI455X (gfx1250) — matrix operand order repaired and checked by reading, not compiled, not run on hardware
//
#include <hip/hip_runtime.h>

// ---------------------------------------------------------------------------
// MHA for MI455X (gfx1250): f16 WMMA pipeline, flash-style attention.
// B=2, N=2048, D=1024, H=16, DH=64.
// ---------------------------------------------------------------------------

typedef _Float16 f16;
typedef __attribute__((ext_vector_type(16))) _Float16 v16h;
typedef __attribute__((ext_vector_type(8)))  _Float16 v8h;
typedef __attribute__((ext_vector_type(8)))  float    v8f;
typedef __attribute__((ext_vector_type(4)))  float    v4f;
typedef __attribute__((ext_vector_type(4)))  _Float16 v4h;

#define DEV static __device__ __forceinline__

constexpr int BB  = 2;
constexpr int NN  = 2048;
constexpr int DD  = 1024;
constexpr int HH  = 16;
constexpr int DHD = 64;
constexpr int MT  = BB * NN;   // 4096 rows total

DEV v8h ld8(const f16* p) { return *(const v8h*)p; }

DEV v16h cat8(v8h a, v8h b) {
  return __builtin_shufflevector(a, b, 0,1,2,3,4,5,6,7,8,9,10,11,12,13,14,15);
}

DEV v8f wmma_f16(v16h a, v16h b, v8f c) {
  // D = A*B + C, 16x16x32, f32 accum
  return __builtin_amdgcn_wmma_f32_16x16x32_f16(false, a, false, b,
                                                (short)0, c, false, false);
}

// ---------------------------------------------------------------------------
// fp32 -> f16 conversion, 4 elements/thread
// ---------------------------------------------------------------------------
__global__ void cvt_f32_f16(const float* __restrict__ in,
                            f16* __restrict__ out, int n4) {
  int i = blockIdx.x * blockDim.x + threadIdx.x;
  if (i < n4) {
    v4f x = ((const v4f*)in)[i];
    v4h y;
    y.x = (f16)x.x; y.y = (f16)x.y; y.z = (f16)x.z; y.w = (f16)x.w;
    ((v4h*)out)[i] = y;
  }
}

// ---------------------------------------------------------------------------
// GEMM: C[m,n] = sum_k A[m,k] * Bw[n,k] + bias[n]   (i.e. y = x * W^T + b)
// A:  [Mtot, K] row-major f16;  Bw: [Nn, K] row-major f16
// One wave computes a 32x64 strip (2 M-tiles x 4 N-tiles), software-pipelined:
// next k-step's 6 operand tiles prefetched into distinct registers while the
// current 8 WMMAs execute.
// mode 0: store f16 row-major [Mtot, Nn]
// mode 1: store f16 V-transposed: out[(m/NN)*DD*NN + n*NN + (m%NN)]
// mode 2: store f32 row-major [Mtot, Nn]
// ---------------------------------------------------------------------------
__global__ __launch_bounds__(128)
void gemm_f16(const f16* __restrict__ A, const f16* __restrict__ Bw,
              const float* __restrict__ bias, void* __restrict__ outp,
              int Mtot, int Nn, int K, int mode) {
  const int tid  = threadIdx.x;
  const int widx = tid >> 5;
  const int lane = tid & 31;
  const int hf   = lane >> 4;   // which 16-lane half
  const int l15  = lane & 15;

  const int gw   = blockIdx.x * 4 + widx;       // global wave index
  const int ntn  = Nn >> 6;                     // n-strips of 64
  const int m0   = (gw / ntn) * 32;
  const int n0   = (gw % ntn) * 64;

  v8f c[2][4];
#pragma unroll
  for (int mt = 0; mt < 2; ++mt)
#pragma unroll
    for (int t = 0; t < 4; ++t)
#pragma unroll
      for (int v = 0; v < 8; ++v) c[mt][t][v] = 0.0f;

  // A-operand (16x32 f16): lanes 0-15 row m hold k 0..15;
  // lanes 16-31 hold k 16..31.
  const f16* arow0 = A + (size_t)(m0 + l15) * K + (hf ? 16 : 0);
  const f16* arow1 = arow0 + (size_t)16 * K;
  // B-operand (32x16 f16): lanes 0-15 col n hold k 0..15; lanes 16-31 k 16..31
  const f16* brow0 = Bw + (size_t)(n0 + l15) * K + (hf ? 16 : 0);
  const f16* brow1 = brow0 + (size_t)16 * K;
  const f16* brow2 = brow0 + (size_t)32 * K;
  const f16* brow3 = brow0 + (size_t)48 * K;

  // prologue: k-step 0
  v16h a0 = cat8(ld8(arow0), ld8(arow0 + 8));
  v16h a1 = cat8(ld8(arow1), ld8(arow1 + 8));
  v16h b0 = cat8(ld8(brow0), ld8(brow0 + 8));
  v16h b1 = cat8(ld8(brow1), ld8(brow1 + 8));
  v16h b2 = cat8(ld8(brow2), ld8(brow2 + 8));
  v16h b3 = cat8(ld8(brow3), ld8(brow3 + 8));

  for (int k0 = 32; k0 < K; k0 += 32) {
    // prefetch next k-step (distinct registers; wait sinks past the WMMAs)
    v16h na0 = cat8(ld8(arow0 + k0), ld8(arow0 + k0 + 8));
    v16h na1 = cat8(ld8(arow1 + k0), ld8(arow1 + k0 + 8));
    v16h nb0 = cat8(ld8(brow0 + k0), ld8(brow0 + k0 + 8));
    v16h nb1 = cat8(ld8(brow1 + k0), ld8(brow1 + k0 + 8));
    v16h nb2 = cat8(ld8(brow2 + k0), ld8(brow2 + k0 + 8));
    v16h nb3 = cat8(ld8(brow3 + k0), ld8(brow3 + k0 + 8));

    c[0][0] = wmma_f16(a0, b0, c[0][0]);
    c[0][1] = wmma_f16(a0, b1, c[0][1]);
    c[0][2] = wmma_f16(a0, b2, c[0][2]);
    c[0][3] = wmma_f16(a0, b3, c[0][3]);
    c[1][0] = wmma_f16(a1, b0, c[1][0]);
    c[1][1] = wmma_f16(a1, b1, c[1][1]);
    c[1][2] = wmma_f16(a1, b2, c[1][2]);
    c[1][3] = wmma_f16(a1, b3, c[1][3]);

    a0 = na0; a1 = na1; b0 = nb0; b1 = nb1; b2 = nb2; b3 = nb3;
  }
  // epilogue: last k-step
  c[0][0] = wmma_f16(a0, b0, c[0][0]);
  c[0][1] = wmma_f16(a0, b1, c[0][1]);
  c[0][2] = wmma_f16(a0, b2, c[0][2]);
  c[0][3] = wmma_f16(a0, b3, c[0][3]);
  c[1][0] = wmma_f16(a1, b0, c[1][0]);
  c[1][1] = wmma_f16(a1, b1, c[1][1]);
  c[1][2] = wmma_f16(a1, b2, c[1][2]);
  c[1][3] = wmma_f16(a1, b3, c[1][3]);

  // C layout: VGPR v, lane half hf -> row = m0 + mt*16 + hf*8 + v, col = n0 + t*16 + l15
#pragma unroll
  for (int mt = 0; mt < 2; ++mt) {
#pragma unroll
    for (int t = 0; t < 4; ++t) {
      const int col = n0 + t * 16 + l15;
      const float bv = bias[col];
#pragma unroll
      for (int v = 0; v < 8; ++v) {
        const int m = m0 + mt * 16 + hf * 8 + v;
        const float val = c[mt][t][v] + bv;
        if (mode == 0) {
          ((f16*)outp)[(size_t)m * Nn + col] = (f16)val;
        } else if (mode == 1) {
          ((f16*)outp)[(size_t)(m >> 11) * DD * NN + (size_t)col * NN + (m & (NN - 1))] = (f16)val;
        } else {
          ((float*)outp)[(size_t)m * Nn + col] = val;
        }
      }
    }
  }
}

// ---------------------------------------------------------------------------
// Flash attention: one wave per (batch, head, 16-query tile).
// Q, Km: f16 [B*N, D] row-major;  Vt: f16 [B][D][N] (key-contiguous rows)
// X out: f16 [B*N, D] row-major.
// Streams keys in chunks of 32 with online softmax.
//  - K tiles are pipelined one chunk ahead (loads issued a full iteration
//    before their S-WMMAs consume them; last iteration wraps harmlessly).
//  - V tiles are issued at the top of the chunk; their latency hides behind
//    the 4 S-WMMAs + the softmax VALU block.
// ---------------------------------------------------------------------------
__global__ __launch_bounds__(128)
void attn_f16(const f16* __restrict__ Q, const f16* __restrict__ Km,
              const f16* __restrict__ Vt, f16* __restrict__ X) {
  __shared__ f16 Pl[4][16 * 32];   // per-wave P tile (16 q x 32 keys)

  const int tid  = threadIdx.x;
  const int widx = tid >> 5;
  const int lane = tid & 31;
  const int hf   = lane >> 4;
  const int l15  = lane & 15;

  const int task = blockIdx.x * 4 + widx;       // B*H*(N/16) = 4096 tasks
  const int bb   = task >> 11;                  // H*(N/16) = 2048 per batch
  const int rem  = task & 2047;
  const int h    = rem >> 7;                    // N/16 = 128 q-tiles per head
  const int qt   = rem & 127;

  // Load Q tile (16 x 64) as two A-operands, kept in registers.
  const f16* qrow = Q + (size_t)(bb * NN + qt * 16 + l15) * DD + h * DHD + (hf ? 16 : 0);
  const v16h aq0 = cat8(ld8(qrow),      ld8(qrow + 8));
  const v16h aq1 = cat8(ld8(qrow + 32), ld8(qrow + 40));

  v8f c[4];
  float mi[8], li[8];
#pragma unroll
  for (int t = 0; t < 4; ++t)
#pragma unroll
    for (int v = 0; v < 8; ++v) c[t][v] = 0.0f;
#pragma unroll
  for (int v = 0; v < 8; ++v) { mi[v] = -1e30f; li[v] = 0.0f; }

  const f16* kbase = Km + (size_t)bb * NN * DD + h * DHD;
  const f16* vbase = Vt + (size_t)bb * DD * NN + (size_t)(h * DHD) * NN;

  // prologue: K tiles for chunk 0
  const f16* kp0 = kbase + (size_t)(l15) * DD + (hf ? 16 : 0);
  const f16* kp1 = kbase + (size_t)(16 + l15) * DD + (hf ? 16 : 0);
  v16h kb00 = cat8(ld8(kp0),      ld8(kp0 + 8));
  v16h kb01 = cat8(ld8(kp0 + 32), ld8(kp0 + 40));
  v16h kb10 = cat8(ld8(kp1),      ld8(kp1 + 8));
  v16h kb11 = cat8(ld8(kp1 + 32), ld8(kp1 + 40));

  for (int kc = 0; kc < NN; kc += 32) {
    // ---- V tiles for this chunk (consumed after softmax) ----
    const f16* vp0 = vbase + (size_t)(l15)      * NN + kc + (hf ? 16 : 0);
    const f16* vp1 = vbase + (size_t)(16 + l15) * NN + kc + (hf ? 16 : 0);
    const f16* vp2 = vbase + (size_t)(32 + l15) * NN + kc + (hf ? 16 : 0);
    const f16* vp3 = vbase + (size_t)(48 + l15) * NN + kc + (hf ? 16 : 0);
    v16h vb0 = cat8(ld8(vp0), ld8(vp0 + 8));
    v16h vb1 = cat8(ld8(vp1), ld8(vp1 + 8));
    v16h vb2 = cat8(ld8(vp2), ld8(vp2 + 8));
    v16h vb3 = cat8(ld8(vp3), ld8(vp3 + 8));

    // ---- prefetch K tiles for the next chunk (wrap keeps loop branch-free) ----
    const int kn = (kc + 32) & (NN - 1);
    const f16* nkp0 = kbase + (size_t)(kn + l15) * DD + (hf ? 16 : 0);
    const f16* nkp1 = kbase + (size_t)(kn + 16 + l15) * DD + (hf ? 16 : 0);
    v16h nk00 = cat8(ld8(nkp0),      ld8(nkp0 + 8));
    v16h nk01 = cat8(ld8(nkp0 + 32), ld8(nkp0 + 40));
    v16h nk10 = cat8(ld8(nkp1),      ld8(nkp1 + 8));
    v16h nk11 = cat8(ld8(nkp1 + 32), ld8(nkp1 + 40));

    // ---- S = Q * K^T (K tiles were loaded one full iteration ago) ----
    v8f s[2];
#pragma unroll
    for (int t = 0; t < 2; ++t)
#pragma unroll
      for (int v = 0; v < 8; ++v) s[t][v] = 0.0f;
    s[0] = wmma_f16(aq0, kb00, s[0]);
    s[0] = wmma_f16(aq1, kb01, s[0]);
    s[1] = wmma_f16(aq0, kb10, s[1]);
    s[1] = wmma_f16(aq1, kb11, s[1]);

    // ---- online softmax per C-row (row lives across a 16-lane half) ----
#pragma unroll
    for (int v = 0; v < 8; ++v) {
      float s0 = s[0][v] * 0.125f;   // 1/sqrt(64)
      float s1 = s[1][v] * 0.125f;
      float rmax = fmaxf(s0, s1);
#pragma unroll
      for (int o = 1; o < 16; o <<= 1) rmax = fmaxf(rmax, __shfl_xor(rmax, o, 16));
      const float mnew  = fmaxf(mi[v], rmax);
      const float alpha = __expf(mi[v] - mnew);
      const float p0 = __expf(s0 - mnew);
      const float p1 = __expf(s1 - mnew);
      float rs = p0 + p1;
#pragma unroll
      for (int o = 1; o < 16; o <<= 1) rs += __shfl_xor(rs, o, 16);
      li[v] = li[v] * alpha + rs;
      mi[v] = mnew;
      c[0][v] *= alpha; c[1][v] *= alpha; c[2][v] *= alpha; c[3][v] *= alpha;
      const int r = hf * 8 + v;
      Pl[widx][r * 32 + l15]      = (f16)p0;
      Pl[widx][r * 32 + 16 + l15] = (f16)p1;
    }
    asm volatile("s_wait_dscnt 0x0" ::: "memory");   // RAW: stores -> A-layout loads

    // ---- reload P (16x32) as A-operand ----
    const f16* pp = &Pl[widx][l15 * 32 + (hf ? 16 : 0)];
    v16h pa = cat8(ld8(pp), ld8(pp + 8));

    // ---- O += P * V  (V tiles issued before the softmax) ----
    c[0] = wmma_f16(pa, vb0, c[0]);
    c[1] = wmma_f16(pa, vb1, c[1]);
    c[2] = wmma_f16(pa, vb2, c[2]);
    c[3] = wmma_f16(pa, vb3, c[3]);
    asm volatile("" ::: "memory");   // keep next iter's P stores after these loads

    kb00 = nk00; kb01 = nk01; kb10 = nk10; kb11 = nk11;
  }

  // ---- normalize and store ----
#pragma unroll
  for (int t = 0; t < 4; ++t) {
#pragma unroll
    for (int v = 0; v < 8; ++v) {
      const int m = bb * NN + qt * 16 + hf * 8 + v;
      X[(size_t)m * DD + h * DHD + t * 16 + l15] = (f16)(c[t][v] / li[v]);
    }
  }
}

// ---------------------------------------------------------------------------
// Host launcher
// ---------------------------------------------------------------------------
extern "C" void kernel_launch(void* const* d_in, const int* in_sizes, int n_in,
                              void* d_out, int out_size, void* d_ws, size_t ws_size,
                              hipStream_t stream) {
  const float* x_q = (const float*)d_in[0];
  const float* x_k = (const float*)d_in[1];
  const float* x_v = (const float*)d_in[2];
  const float* Wq  = (const float*)d_in[3];
  const float* bq  = (const float*)d_in[4];
  const float* Wk  = (const float*)d_in[5];
  const float* bk  = (const float*)d_in[6];
  const float* Wv  = (const float*)d_in[7];
  const float* bv  = (const float*)d_in[8];
  const float* Wo  = (const float*)d_in[9];
  const float* bo  = (const float*)d_in[10];

  const size_t XE = (size_t)MT * DD;   // 4M elements per activation tensor
  const size_t WE = (size_t)DD * DD;   // 1M elements per weight

  f16* ws   = (f16*)d_ws;
  f16* xq16 = ws;
  f16* xk16 = xq16 + XE;
  f16* xv16 = xk16 + XE;
  f16* wq16 = xv16 + XE;
  f16* wk16 = wq16 + WE;
  f16* wv16 = wk16 + WE;
  f16* wo16 = wv16 + WE;
  f16* q16  = wo16 + WE;
  f16* k16  = q16 + XE;
  f16* vt16 = k16 + XE;
  f16* x16  = vt16 + XE;
  // total: 33,554,432 f16 = 64 MB of workspace

  auto cvt = [&](const float* src, f16* dst, size_t n) {
    int n4 = (int)(n / 4);
    cvt_f32_f16<<<(n4 + 255) / 256, 256, 0, stream>>>(src, dst, n4);
  };
  cvt(x_q, xq16, XE);
  cvt(x_k, xk16, XE);
  cvt(x_v, xv16, XE);
  cvt(Wq, wq16, WE);
  cvt(Wk, wk16, WE);
  cvt(Wv, wv16, WE);
  cvt(Wo, wo16, WE);

  // (MT/32)*(DD/64) = 2048 wave strips / 4 waves per block = 512 blocks
  const int gemm_blocks = (MT / 32) * (DD / 64) / 4;
  gemm_f16<<<gemm_blocks, 128, 0, stream>>>(xq16, wq16, bq, q16,  MT, DD, DD, 0);
  gemm_f16<<<gemm_blocks, 128, 0, stream>>>(xk16, wk16, bk, k16,  MT, DD, DD, 0);
  gemm_f16<<<gemm_blocks, 128, 0, stream>>>(xv16, wv16, bv, vt16, MT, DD, DD, 1);

  // B*H*(N/16) = 4096 wave tasks / 4 per block = 1024 blocks
  const int attn_blocks = BB * HH * (NN / 16) / 4;
  attn_f16<<<attn_blocks, 128, 0, stream>>>(q16, k16, vt16, x16);

  gemm_f16<<<gemm_blocks, 128, 0, stream>>>(x16, wo16, bo, d_out, MT, DD, DD, 2);
}
